// VectorizedPinSAGELayer_2353642078648
// MI455X (gfx1250) — matrix operand order repaired and checked by reading, not compiled, not run on hardware
//
#include <hip/hip_runtime.h>
#include <hip/hip_bf16.h>

// ---------------------------------------------------------------------------
// PinSAGE layer, MI455X (gfx1250, wave32, WMMA).
//   out = relu( (W_n·h_nbr + b_n + W_s·x_self + b_s) · W_c^T + b_c )
// Gather phase is HBM-bandwidth bound (~885MB @ 23.3TB/s ≈ 38us floor);
// GEMMs run on v_wmma_f32_16x16x32_bf16 (f32 accumulate) and hide under it.
// ---------------------------------------------------------------------------

typedef __bf16 bf16_t;
typedef bf16_t v16bf __attribute__((ext_vector_type(16)));
typedef float  v8f   __attribute__((ext_vector_type(8)));

#define DIMS   128
#define KNBR   32
#define TILE_M 16

// --- prep: convert the three 128x128 f32 weight matrices to bf16 in ws -----
__global__ void cvt_weights_bf16(const float* __restrict__ wn,
                                 const float* __restrict__ ws,
                                 const float* __restrict__ wc,
                                 bf16_t* __restrict__ out) {
    int i = blockIdx.x * blockDim.x + threadIdx.x;     // 0 .. 3*128*128
    const int E = DIMS * DIMS;
    if (i < E)            out[i] = (bf16_t)wn[i];
    else if (i < 2 * E)   out[i] = (bf16_t)ws[i - E];
    else if (i < 3 * E)   out[i] = (bf16_t)wc[i - 2 * E];
}

// --- main: one 4-wave workgroup handles a 16-row tile ----------------------
__launch_bounds__(128)
__global__ void pinsage_tile(const int*   __restrict__ node_ids,
                             const float* __restrict__ emb,
                             const int*   __restrict__ nbr_ids,
                             const float* __restrict__ nbr_w,
                             const bf16_t* __restrict__ Wn,   // [128][128] bf16
                             const bf16_t* __restrict__ Ws,
                             const bf16_t* __restrict__ Wc,
                             const float* __restrict__ bn,
                             const float* __restrict__ bs,
                             const float* __restrict__ bc,
                             float* __restrict__ out,
                             int B) {
    __shared__ float  s_wt[TILE_M][DIMS + 4];   // weighted neighbor sum (f32)
    __shared__ float  s_xs[TILE_M][DIMS + 4];   // self embedding        (f32)
    __shared__ bf16_t s_hid[TILE_M][DIMS + 8];  // hidden (bf16, 2nd GEMM A)

    const int tid  = threadIdx.x;
    const int lane = tid & 31;        // wave32
    const int wv   = tid >> 5;        // 0..3
    const int b0   = blockIdx.x * TILE_M;

    // ---------------- gather + weighted neighbor reduce --------------------
    // One wave per node-row: 32 lanes * float4 = one full 512B emb row.
    const float4* emb4 = (const float4*)emb;
    for (int ml = 0; ml < 4; ++ml) {
        const int row = wv * 4 + ml;
        int b = b0 + row;
        if (b >= B) b = B - 1;                        // clamp (tail tile)
        const int nid = node_ids[b];

        const int   id_l = nbr_ids[(size_t)nid * KNBR + lane];  // coalesced
        const float w_l  = nbr_w [(size_t)nid * KNBR + lane];
        const float4 xs  = emb4[(size_t)nid * (DIMS / 4) + lane];

        float4 acc = make_float4(0.f, 0.f, 0.f, 0.f);
        #pragma unroll 8
        for (int k = 0; k < KNBR; ++k) {
            const int   idk = __shfl(id_l, k, 32);    // broadcast nbr id
            const float wk  = __shfl(w_l,  k, 32);    // broadcast nbr weight
            const float4 e  = emb4[(size_t)idk * (DIMS / 4) + lane];
            acc.x += wk * e.x; acc.y += wk * e.y;
            acc.z += wk * e.z; acc.w += wk * e.w;
        }
        float* pw = &s_wt[row][lane * 4];
        pw[0] = acc.x; pw[1] = acc.y; pw[2] = acc.z; pw[3] = acc.w;
        float* px = &s_xs[row][lane * 4];
        px[0] = xs.x;  px[1] = xs.y;  px[2] = xs.z;  px[3] = xs.w;
    }
    __syncthreads();

    const int m   = lane & 15;        // row (A) / col-in-block (B,C,D)
    const int hal = lane >> 4;        // half-wave select

    // -------- GEMM 1: hidden = weighted*Wn^T + x_self*Ws^T + (bn+bs) -------
    // Single f32 accumulator over a fused K=256 reduction (Wn then Ws).
    #pragma unroll
    for (int nb = 0; nb < 2; ++nb) {
        const int nblk = wv * 2 + nb;            // 8 N-blocks over 4 waves
        const int col  = nblk * 16 + m;
        v8f acc = {0.f, 0.f, 0.f, 0.f, 0.f, 0.f, 0.f, 0.f};

        #pragma unroll
        for (int src = 0; src < 2; ++src) {
            const float*  sA = src ? &s_xs[m][0] : &s_wt[m][0];
            const bf16_t* W  = src ? Ws : Wn;
            #pragma unroll
            for (int kk = 0; kk < 4; ++kk) {
                const int kb = kk * 32 + hal * 8;        // A frag K base
                v16bf a, bfr;
                #pragma unroll
                for (int i = 0; i < 8; ++i) {            // ISA 16-bit A layout
                    a[i]     = (bf16_t)sA[kb + i];       // K = kb..kb+7
                    a[8 + i] = (bf16_t)sA[kb + 16 + i];  // K = kb+16..kb+23
                }
                const bf16_t* p = W + (size_t)col * DIMS + kk * 32 + hal * 8;
                #pragma unroll
                for (int i = 0; i < 16; ++i) bfr[i] = p[i < 8 ? i : i + 8];  // K = kb..kb+7, kb+16..kb+23
                acc = __builtin_amdgcn_wmma_f32_16x16x32_bf16(
                          false, a, false, bfr, (short)0, acc, false, false);
            }
        }
        const float bias = bn[col] + bs[col];
        #pragma unroll
        for (int i = 0; i < 8; ++i)                     // C/D: row=i+8*hal, N=m
            s_hid[i + hal * 8][col] = (bf16_t)(acc[i] + bias);
    }
    __syncthreads();

    // -------- GEMM 2: out = relu(hidden * Wc^T + bc) ------------------------
    #pragma unroll
    for (int nb = 0; nb < 2; ++nb) {
        const int nblk = wv * 2 + nb;
        const int col  = nblk * 16 + m;
        v8f acc = {0.f, 0.f, 0.f, 0.f, 0.f, 0.f, 0.f, 0.f};

        #pragma unroll
        for (int kk = 0; kk < 4; ++kk) {
            const int kb = kk * 32 + hal * 8;
            v16bf a, bfr;
            const bf16_t* r = &s_hid[m][0];
            #pragma unroll
            for (int i = 0; i < 8; ++i) {
                a[i]     = r[kb + i];
                a[8 + i] = r[kb + 16 + i];
            }
            const bf16_t* p = Wc + (size_t)col * DIMS + kk * 32 + hal * 8;
            #pragma unroll
            for (int i = 0; i < 16; ++i) bfr[i] = p[i < 8 ? i : i + 8];
            acc = __builtin_amdgcn_wmma_f32_16x16x32_bf16(
                      false, a, false, bfr, (short)0, acc, false, false);
        }
        const float bias = bc[col];
        #pragma unroll
        for (int i = 0; i < 8; ++i) {
            const int rowg = b0 + i + hal * 8;
            if (rowg < B)
                out[(size_t)rowg * DIMS + col] = fmaxf(acc[i] + bias, 0.f);
        }
    }
}

// ---------------------------------------------------------------------------
extern "C" void kernel_launch(void* const* d_in, const int* in_sizes, int n_in,
                              void* d_out, int out_size, void* d_ws, size_t ws_size,
                              hipStream_t stream) {
    (void)n_in; (void)out_size; (void)ws_size;
    const int*   node_ids = (const int*)  d_in[0];
    const float* emb      = (const float*)d_in[1];
    const int*   nbr_ids  = (const int*)  d_in[2];
    const float* nbr_w    = (const float*)d_in[3];
    const float* Wn_w     = (const float*)d_in[4];
    const float* Wn_b     = (const float*)d_in[5];
    const float* Ws_w     = (const float*)d_in[6];
    const float* Ws_b     = (const float*)d_in[7];
    const float* Wc_w     = (const float*)d_in[8];
    const float* Wc_b     = (const float*)d_in[9];
    float* out = (float*)d_out;
    const int B = in_sizes[0];

    bf16_t* wbf = (bf16_t*)d_ws;                 // 3 * 128*128 bf16 = 96 KB
    const int E = DIMS * DIMS;
    cvt_weights_bf16<<<(3 * E + 255) / 256, 256, 0, stream>>>(Wn_w, Ws_w, Wc_w, wbf);

    const int tiles = (B + TILE_M - 1) / TILE_M;
    pinsage_tile<<<tiles, 128, 0, stream>>>(node_ids, emb, nbr_ids, nbr_w,
                                            wbf, wbf + E, wbf + 2 * E,
                                            Wn_b, Ws_b, Wc_b, out, B);
}
